// SPEncoder_88390426951914
// MI455X (gfx1250) — matrix operand order repaired and checked by reading, not compiled, not run on hardware
//
#include <hip/hip_runtime.h>
#include <hip/hip_bf16.h>
#include <stdint.h>

#define B_  4
#define N_  4096
#define D_  1024
#define H_  16
#define DH_ 64
#define M_  256

static constexpr float DN    = 0.35355339059327373f; // DH^-0.25
static constexpr float RATIO = 0.0625f;              // M^-0.5
static constexpr float EPSV  = 1e-4f;

typedef __attribute__((ext_vector_type(16))) __bf16 v16bf;
typedef __attribute__((ext_vector_type(8)))  float  v8f;

union BF16Frag { v16bf v; unsigned u[8]; };

__device__ __forceinline__ unsigned short f2bf(float f) {
    unsigned x = __float_as_uint(f);
    x += 0x7fffu + ((x >> 16) & 1u);           // round-to-nearest-even
    return (unsigned short)(x >> 16);
}
__device__ __forceinline__ float bf2f(unsigned h) { return __uint_as_float(h << 16); }
__device__ __forceinline__ int   fenc(float f) { int b = __float_as_int(f); return b >= 0 ? b : (b ^ 0x7fffffff); }
__device__ __forceinline__ float fdec(int k)   { int b = k >= 0 ? k : (k ^ 0x7fffffff); return __int_as_float(b); }

// ---------------------------------------------------------------------------
// Tensor Data Mover support (gfx1250). D# layout per CDNA5 ISA §8.3/8.4.
// This toolchain exposes the 6-arg builtin:
//   (u32x4 g0, i32x8 g1, i32x4 g2, i32x4 g3, i32x8 g4, i32 cpol)
// ---------------------------------------------------------------------------
#ifndef __has_builtin
#define __has_builtin(x) 0
#endif
#if __has_builtin(__builtin_amdgcn_tensor_load_to_lds) && \
    __has_builtin(__builtin_amdgcn_s_wait_tensorcnt)
#define HAVE_TDM 1
#else
#define HAVE_TDM 0
#endif

#if HAVE_TDM
typedef unsigned u32x4 __attribute__((ext_vector_type(4)));
typedef int      i32x8 __attribute__((ext_vector_type(8)));
typedef int      i32x4 __attribute__((ext_vector_type(4)));

// DMA a 2D tile of bf16 (tile_k elems per row, tile_rows rows, row stride
// row_stride elems) from global to LDS, inserting 4 pad DWORDs after every
// 16 data DWORDs (64B row -> 80B LDS stride, matches Asm[row*20] layout).
__device__ __forceinline__ void tdm_load_tile_2d(const void* gptr, unsigned lds_off,
                                                 int tile_k, int tile_rows,
                                                 int row_stride) {
    unsigned long long ga = (unsigned long long)(uintptr_t)gptr;
    u32x4 g0;
    g0[0] = 1u;                                           // count=1, user D#
    g0[1] = lds_off;                                      // lds_addr (bytes)
    g0[2] = (unsigned)(ga & 0xffffffffu);                 // global_addr[31:0]
    g0[3] = (unsigned)((ga >> 32) & 0x01ffffffu)          // global_addr[56:32]
          | (2u << 30);                                   // type=2 ("image")
    i32x8 g1;
    g1[0] = (int)((1u << 16)        // data_size = 2 bytes
                | (1u << 20)        // pad_enable
                | (3u << 22)        // pad_interval: 16 DWORDs (64B)
                | (3u << 25));      // pad_amount: 4 DWORDs (16B)
    g1[1] = 0;                                            // abar=0, dim0[15:0]=0
    g1[2] = (int)0x4000;                                  // dim0=1<<30, dim1 lo=0
    g1[3] = (int)(0x4000u | ((unsigned)tile_k << 16));    // dim1=1<<30, tile_dim0
    g1[4] = (int)((unsigned)tile_rows & 0xffffu);         // tile_dim1, tile_dim2=0
    g1[5] = row_stride;                                   // dim0_stride[31:0]
    g1[6] = 0;                                            // dim0_stride hi, dim1_stride lo
    g1[7] = 0;
    i32x4 gz4 = {0, 0, 0, 0};
    i32x8 gz8 = {0, 0, 0, 0, 0, 0, 0, 0};
    __builtin_amdgcn_tensor_load_to_lds(g0, g1, gz4, gz4, gz8, 0);
}
#endif

// ---------------------------------------------------------------------------
// fp32 -> bf16 conversion (vectorized, optional scale)
// ---------------------------------------------------------------------------
__global__ void cvt_kernel(const float* __restrict__ src, unsigned short* __restrict__ dst,
                           float scale, int n4) {
    int i = blockIdx.x * blockDim.x + threadIdx.x;
    if (i >= n4) return;
    float4 f = ((const float4*)src)[i];
    unsigned lo = (unsigned)f2bf(f.x * scale) | ((unsigned)f2bf(f.y * scale) << 16);
    unsigned hi = (unsigned)f2bf(f.z * scale) | ((unsigned)f2bf(f.w * scale) << 16);
    ((uint2*)dst)[i] = make_uint2(lo, hi);
}

__global__ void init_kernel(float* __restrict__ csum, int* __restrict__ gmax) {
    int i = blockIdx.x * blockDim.x + threadIdx.x;
    for (int j = i; j < B_ * H_ * M_; j += blockDim.x * gridDim.x) csum[j] = 0.0f;
    if (i == 0) *gmax = (int)0x80000000;       // < fenc of any float
}

// ---------------------------------------------------------------------------
// Generic batched bf16 GEMM, C = A(MxK) * B(KxN), fp32 accumulate via WMMA.
// Block = 128 threads (4 waves). Tile 128x64, K-step 32; each wave owns a
// 32x64 sub-tile (2 A frags x 4 B frags = 8 v_wmma per k-step).
// A tile is staged with the Tensor Data Mover (one 32-row DMA per wave,
// padded directly into the conflict-free [row][20 u32] LDS layout); the
// B tile needs a K-pair interleave transform, so it stays on the
// load->pack->ds_store path.  Fragments load as two ds_load_b128 each.
// ---------------------------------------------------------------------------
template <bool TRANS_A, bool OUT_BF16, bool HAS_BIAS, bool HAS_RS>
__global__ void __launch_bounds__(128)
gemm_bf16(const unsigned short* __restrict__ A, long long aS0, long long aS1, int lda,
          const unsigned short* __restrict__ Bg, long long bS0, long long bS1, int ldb,
          void* __restrict__ Cg, long long cS0, long long cS1, int ldc,
          const float* __restrict__ bias,
          const float* __restrict__ rs, long long rS0, long long rS1,
          int Kdim) {
    const int t = threadIdx.x, wave = t >> 5, lane = t & 31;
    const int row16 = lane & 15, lh = lane >> 4;
    const int z = blockIdx.z, zb = z >> 4, zh = z & 15;
    const int m0 = blockIdx.x * 128, n0 = blockIdx.y * 64;

    const unsigned short* Ab = A + (long long)zb * aS0 + (long long)zh * aS1;
    const unsigned short* Bb = Bg + (long long)zb * bS0 + (long long)zh * bS1;

    __shared__ unsigned Asm[128 * 20];
    __shared__ unsigned Bsm[64 * 20];

    v8f acc[2][4] = {};

    for (int k0 = 0; k0 < Kdim; k0 += 32) {
        // ---- stage A tile (128 rows x 32 k) ----
        if constexpr (!TRANS_A) {
#if HAVE_TDM
            // one TDM 2D-tile DMA per wave: rows [wave*32, wave*32+32)
            tdm_load_tile_2d(Ab + (size_t)(m0 + wave * 32) * lda + k0,
                             (unsigned)(uintptr_t)(void*)Asm + (unsigned)(wave * 32) * 80u,
                             32, 32, lda);
#else
            const uint4* src = (const uint4*)(Ab + (size_t)(m0 + t) * lda + k0);
            uint4 a0 = src[0], a1 = src[1], a2 = src[2], a3 = src[3];
            uint4* dst = (uint4*)&Asm[t * 20];
            dst[0] = a0; dst[1] = a1; dst[2] = a2; dst[3] = a3;
            if (k0 + 32 < Kdim)
                __builtin_prefetch(Ab + (size_t)(m0 + t) * lda + k0 + 32, 0, 3);
#endif
        } else {
            // A element (m,k) lives at Ab[k*lda + m]; transpose into LDS
            const int kk = t & 31, mseg = t >> 5;   // mseg covers 32 cols
            const unsigned short* src = Ab + (size_t)(k0 + kk) * lda + m0 + mseg * 32;
            uint4 a0 = ((const uint4*)src)[0], a1 = ((const uint4*)src)[1];
            uint4 a2 = ((const uint4*)src)[2], a3 = ((const uint4*)src)[3];
            const unsigned short* e0 = (const unsigned short*)&a0;
            const unsigned short* e1 = (const unsigned short*)&a1;
            const unsigned short* e2 = (const unsigned short*)&a2;
            const unsigned short* e3 = (const unsigned short*)&a3;
            unsigned short* As16 = (unsigned short*)Asm;
#pragma unroll
            for (int j = 0; j < 8; ++j) As16[(mseg * 32 + j) * 40 + kk]      = e0[j];
#pragma unroll
            for (int j = 0; j < 8; ++j) As16[(mseg * 32 + 8 + j) * 40 + kk]  = e1[j];
#pragma unroll
            for (int j = 0; j < 8; ++j) As16[(mseg * 32 + 16 + j) * 40 + kk] = e2[j];
#pragma unroll
            for (int j = 0; j < 8; ++j) As16[(mseg * 32 + 24 + j) * 40 + kk] = e3[j];
            if (k0 + 32 < Kdim)
                __builtin_prefetch(Ab + (size_t)(k0 + 32 + kk) * lda + m0 + mseg * 32, 0, 3);
        }
        // ---- stage B tile (32 k x 64 n), fragment-ordered ----
        {
            const int pi = t >> 3, cb = (t & 7) * 8;     // k-pair index, col base
            const unsigned short* r0 = Bb + (size_t)(k0 + 2 * pi) * ldb + n0 + cb;
            uint4 lo = ((const uint4*)r0)[0];
            uint4 hi = ((const uint4*)(r0 + ldb))[0];
            const unsigned short* el = (const unsigned short*)&lo;
            const unsigned short* eh = (const unsigned short*)&hi;
            const int slot = (pi >> 3) * 8 + (pi & 7);   // half*8 + i
#pragma unroll
            for (int j = 0; j < 8; ++j)
                Bsm[(cb + j) * 20 + slot] = (unsigned)el[j] | ((unsigned)eh[j] << 16);
            if (k0 + 32 < Kdim) __builtin_prefetch(r0 + (size_t)32 * ldb, 0, 3);
        }
#if HAVE_TDM
        if constexpr (!TRANS_A) __builtin_amdgcn_s_wait_tensorcnt(0);
#endif
        __syncthreads();

        // ---- A fragments: 16x32 bf16 per ISA layout ----
        BF16Frag af[2];
#pragma unroll
        for (int mi = 0; mi < 2; ++mi)
#pragma unroll
            for (int i = 0; i < 8; ++i) {
                int p = (i < 4 ? 0 : 8) + lh * 4 + (i & 3);
                af[mi].u[i] = Asm[(wave * 32 + mi * 16 + row16) * 20 + p];
            }
        // ---- B fragments: same k-pair order as A fragments ----
#pragma unroll
        for (int nb = 0; nb < 4; ++nb) {
            BF16Frag bfr;
#pragma unroll
            for (int i = 0; i < 8; ++i)
                bfr.u[i] = Bsm[(nb * 16 + row16) * 20 + (i < 4 ? 0 : 8) + lh * 4 + (i & 3)];
#pragma unroll
            for (int mi = 0; mi < 2; ++mi)
                acc[mi][nb] = __builtin_amdgcn_wmma_f32_16x16x32_bf16(
                    false, af[mi].v, false, bfr.v, (short)0, acc[mi][nb], false, false);
        }
        __syncthreads();
    }

    // ---- epilogue ----
    const float* rsb = nullptr;
    if constexpr (HAS_RS) rsb = rs + (long long)zb * rS0 + (long long)zh * rS1;
    long long coff = (long long)zb * cS0 + (long long)zh * cS1;
#pragma unroll
    for (int mi = 0; mi < 2; ++mi) {
#pragma unroll
        for (int nb = 0; nb < 4; ++nb) {
#pragma unroll
            for (int r = 0; r < 8; ++r) {
                int grow = m0 + wave * 32 + mi * 16 + (lh ? r + 8 : r);
                int gcol = n0 + nb * 16 + row16;
                float val = acc[mi][nb][r];
                if constexpr (HAS_RS)   val *= rsb[grow];
                if constexpr (HAS_BIAS) val += bias[gcol];
                size_t idx = (size_t)coff + (size_t)grow * ldc + gcol;
                if constexpr (OUT_BF16) ((unsigned short*)Cg)[idx] = f2bf(val);
                else                    ((float*)Cg)[idx] = val;
            }
        }
    }
}

// ---------------------------------------------------------------------------
// FAVOR+ feature map. Block = 64 threads (2 waves), covers 32 rows x M=256.
// dd = data . (dn*P)^T via WMMA (dn folded into Pb).
// MODE 0: global max of dd (int-encoded atomicMax) -- for keys.
// MODE 1: per-row max, exp, write qp (bf16).
// MODE 2: global max, exp, write kp (bf16) + column sums into k_cumsum.
// ---------------------------------------------------------------------------
template <int MODE>
__global__ void __launch_bounds__(64)
feat_kernel(const unsigned short* __restrict__ data,   // [b,n,h*64+d] bf16
            const unsigned short* __restrict__ Pb,     // [256][64] bf16 (pre-scaled by dn)
            unsigned short* __restrict__ outp,         // [(b*H+h)*N + n][256] bf16
            float* __restrict__ csum,                  // [B*H*256]
            int* __restrict__ gmax) {
    const int t = threadIdx.x, wave = t >> 5, lane = t & 31;
    const int row16 = lane & 15, lh = lane >> 4;
    const int bh = blockIdx.y, b = bh >> 4, h = bh & 15;
    const int n0 = blockIdx.x * 32;

    __shared__ float dd_s[32 * 260];
    __shared__ float red_s[64];
    __shared__ float diag_s[32];
    __shared__ float rmax_s[32];

    // diag partials: 0.5*dn^2 * sum_d data^2
    if constexpr (MODE != 0) {
        int row = t >> 1, seg = t & 1;
        const unsigned* dr = (const unsigned*)data +
            (((size_t)(b * N_ + n0 + row) * 1024 + h * 64) >> 1) + seg * 16;
        float s = 0.0f;
#pragma unroll
        for (int j = 0; j < 16; ++j) {
            unsigned u = dr[j];
            float x0 = bf2f(u & 0xffffu), x1 = bf2f(u >> 16);
            s += x0 * x0 + x1 * x1;
        }
        red_s[t] = s;
        __syncthreads();
        if (t < 32) diag_s[t] = (red_s[2 * t] + red_s[2 * t + 1]) * (0.5f * DN * DN);
    }

    // A fragments straight from global (rows n0+wave*16+row16, K = d 0..63)
    const unsigned* drow = (const unsigned*)data +
        (((size_t)(b * N_ + n0 + wave * 16 + row16) * 1024 + h * 64) >> 1);
    BF16Frag af[2];
#pragma unroll
    for (int f = 0; f < 2; ++f)
#pragma unroll
        for (int i = 0; i < 8; ++i) {
            int p = (i < 4 ? 0 : 8) + lh * 4 + (i & 3);
            af[f].u[i] = drow[f * 16 + p];
        }

    const unsigned* Pu = (const unsigned*)Pb;
    float lmax = -3.4e38f;
    for (int mb = 0; mb < 16; ++mb) {
        v8f acc = {};
#pragma unroll
        for (int f = 0; f < 2; ++f) {
            BF16Frag bfr;
#pragma unroll
            for (int i = 0; i < 8; ++i)
                bfr.u[i] = Pu[(size_t)(mb * 16 + row16) * 32 + f * 16 + (i < 4 ? 0 : 8) + lh * 4 + (i & 3)];
            acc = __builtin_amdgcn_wmma_f32_16x16x32_bf16(
                false, af[f].v, false, bfr.v, (short)0, acc, false, false);
        }
        if constexpr (MODE == 0) {
#pragma unroll
            for (int r = 0; r < 8; ++r) lmax = fmaxf(lmax, acc[r]);
        } else {
#pragma unroll
            for (int r = 0; r < 8; ++r)
                dd_s[(wave * 16 + (lh ? r + 8 : r)) * 260 + mb * 16 + row16] = acc[r];
        }
    }
    __syncthreads();

    if constexpr (MODE == 0) {
        red_s[t] = lmax;
        __syncthreads();
        for (int s = 32; s > 0; s >>= 1) {
            if (t < s) red_s[t] = fmaxf(red_s[t], red_s[t + s]);
            __syncthreads();
        }
        if (t == 0) atomicMax(gmax, fenc(red_s[0]));
    } else {
        const int row = t >> 1, off = (t & 1) * 128;
        float* dr = &dd_s[row * 260 + off];
        float mx;
        if constexpr (MODE == 1) {
            float lm = -3.4e38f;
            for (int j = 0; j < 128; ++j) lm = fmaxf(lm, dr[j]);
            red_s[t] = lm;
            __syncthreads();
            if (t < 32) rmax_s[t] = fmaxf(red_s[2 * t], red_s[2 * t + 1]);
            __syncthreads();
            mx = rmax_s[row];
        } else {
            mx = fdec(*gmax);
        }
        const float dg = diag_s[row];
        unsigned* orow = (unsigned*)(outp + ((size_t)bh * N_ + n0 + row) * 256 + off);
        for (int j = 0; j < 128; j += 2) {
            float v0 = RATIO * (__expf(dr[j]     - dg - mx) + EPSV);
            float v1 = RATIO * (__expf(dr[j + 1] - dg - mx) + EPSV);
            if constexpr (MODE == 2) { dr[j] = v0; dr[j + 1] = v1; }
            orow[j >> 1] = (unsigned)f2bf(v0) | ((unsigned)f2bf(v1) << 16);
        }
        if constexpr (MODE == 2) {
            __syncthreads();
            float* cb = csum + (size_t)bh * 256;
            for (int c = t * 4; c < t * 4 + 4; ++c) {
                float s = 0.0f;
                for (int r = 0; r < 32; ++r) s += dd_s[r * 260 + c];
                atomicAdd(&cb[c], s);
            }
        }
    }
}

// ---------------------------------------------------------------------------
// D_inv[b,h,n] = 1 / (qp[b,h,n,:] . k_cumsum[b,h,:]); one wave per row.
// ---------------------------------------------------------------------------
__global__ void __launch_bounds__(128)
dinv_kernel(const unsigned short* __restrict__ qp, const float* __restrict__ csum,
            float* __restrict__ dinv) {
    const int t = threadIdx.x, wave = t >> 5, lane = t & 31;
    const size_t r = (size_t)blockIdx.x * 4 + wave;    // row in [0, B*H*N)
    const int bh = (int)(r >> 12);                     // N = 4096
    const unsigned* row = (const unsigned*)qp + r * 128;
    const float* cs = csum + (size_t)bh * 256;
    float s = 0.0f;
    for (int j = lane; j < 128; j += 32) {
        unsigned u = row[j];
        s += bf2f(u & 0xffffu) * cs[2 * j] + bf2f(u >> 16) * cs[2 * j + 1];
    }
#pragma unroll
    for (int m = 16; m; m >>= 1) s += __shfl_xor(s, m, 32);
    if (lane == 0) dinv[r] = 1.0f / s;
}

// ---------------------------------------------------------------------------
extern "C" void kernel_launch(void* const* d_in, const int* in_sizes, int n_in,
                              void* d_out, int out_size, void* d_ws, size_t ws_size,
                              hipStream_t stream) {
    (void)in_sizes; (void)n_in; (void)out_size; (void)ws_size;
    const float* x  = (const float*)d_in[0];
    const float* Wq = (const float*)d_in[1];
    const float* Wk = (const float*)d_in[2];
    const float* Wv = (const float*)d_in[3];
    const float* Wo = (const float*)d_in[4];
    const float* bo = (const float*)d_in[5];
    const float* P  = (const float*)d_in[6];

    char* ws = (char*)d_ws;
    size_t off = 0;
    auto carve = [&](size_t bytes) -> char* {
        char* p = ws + off;
        off = (off + bytes + 255) & ~(size_t)255;
        return p;
    };
    unsigned short* xb  = (unsigned short*)carve((size_t)B_ * N_ * D_ * 2);
    unsigned short* wqb = (unsigned short*)carve((size_t)D_ * D_ * 2);
    unsigned short* wkb = (unsigned short*)carve((size_t)D_ * D_ * 2);
    unsigned short* wvb = (unsigned short*)carve((size_t)D_ * D_ * 2);
    unsigned short* wob = (unsigned short*)carve((size_t)D_ * D_ * 2);
    unsigned short* pb  = (unsigned short*)carve((size_t)M_ * DH_ * 2);
    unsigned short* qb  = (unsigned short*)carve((size_t)B_ * N_ * D_ * 2);
    unsigned short* kb  = (unsigned short*)carve((size_t)B_ * N_ * D_ * 2);
    unsigned short* vb  = (unsigned short*)carve((size_t)B_ * N_ * D_ * 2);
    unsigned short* qp  = (unsigned short*)carve((size_t)B_ * H_ * N_ * M_ * 2);
    unsigned short* kp  = (unsigned short*)carve((size_t)B_ * H_ * N_ * M_ * 2);
    float*          cs  = (float*)carve((size_t)B_ * H_ * M_ * 4);
    int*            gm  = (int*)carve(256);
    unsigned short* ctx = (unsigned short*)carve((size_t)B_ * H_ * M_ * DH_ * 2);
    float*          dv  = (float*)carve((size_t)B_ * H_ * N_ * 4);
    unsigned short* at  = (unsigned short*)carve((size_t)B_ * N_ * D_ * 2);

    // conversions + init
    cvt_kernel<<<(B_ * N_ * D_ / 4 + 255) / 256, 256, 0, stream>>>(x, xb, 1.0f, B_ * N_ * D_ / 4);
    cvt_kernel<<<(D_ * D_ / 4 + 255) / 256, 256, 0, stream>>>(Wq, wqb, 1.0f, D_ * D_ / 4);
    cvt_kernel<<<(D_ * D_ / 4 + 255) / 256, 256, 0, stream>>>(Wk, wkb, 1.0f, D_ * D_ / 4);
    cvt_kernel<<<(D_ * D_ / 4 + 255) / 256, 256, 0, stream>>>(Wv, wvb, 1.0f, D_ * D_ / 4);
    cvt_kernel<<<(D_ * D_ / 4 + 255) / 256, 256, 0, stream>>>(Wo, wob, 1.0f, D_ * D_ / 4);
    cvt_kernel<<<(M_ * DH_ / 4 + 255) / 256, 256, 0, stream>>>(P, pb, DN, M_ * DH_ / 4);
    init_kernel<<<1, 256, 0, stream>>>(cs, gm);

    const dim3 blk(128);
    // QKV projections: [16384 x 1024] * [1024 x 1024] -> bf16 [b,n,h*64+d]
    const dim3 gq(128, 16, 1);
    gemm_bf16<false, true, false, false><<<gq, blk, 0, stream>>>(
        xb, 0, 0, 1024, wqb, 0, 0, 1024, qb, 0, 0, 1024, nullptr, nullptr, 0, 0, 1024);
    gemm_bf16<false, true, false, false><<<gq, blk, 0, stream>>>(
        xb, 0, 0, 1024, wkb, 0, 0, 1024, kb, 0, 0, 1024, nullptr, nullptr, 0, 0, 1024);
    gemm_bf16<false, true, false, false><<<gq, blk, 0, stream>>>(
        xb, 0, 0, 1024, wvb, 0, 0, 1024, vb, 0, 0, 1024, nullptr, nullptr, 0, 0, 1024);

    // feature maps
    const dim3 gf(N_ / 32, B_ * H_, 1);
    feat_kernel<0><<<gf, 64, 0, stream>>>(kb, pb, nullptr, nullptr, gm);   // global max(dd_k)
    feat_kernel<1><<<gf, 64, 0, stream>>>(qb, pb, qp, nullptr, gm);        // qp
    feat_kernel<2><<<gf, 64, 0, stream>>>(kb, pb, kp, cs, gm);             // kp + k_cumsum

    dinv_kernel<<<(B_ * H_ * N_) / 4, 128, 0, stream>>>(qp, cs, dv);

    // ctx[b,h] = kp^T (256 x 4096) * v (4096 x 64)
    const dim3 gc(2, 1, 64);
    gemm_bf16<true, true, false, false><<<gc, blk, 0, stream>>>(
        kp, 16LL * N_ * M_, (long long)N_ * M_, M_,
        vb, (long long)N_ * 1024, 64LL, 1024,
        ctx, 16LL * M_ * DH_, (long long)M_ * DH_, DH_,
        nullptr, nullptr, 0, 0, N_);

    // out[b,h] = D_inv * (qp (4096 x 256) * ctx (256 x 64)) -> attn [b,n,h*64+e]
    const dim3 go(32, 1, 64);
    gemm_bf16<false, true, false, true><<<go, blk, 0, stream>>>(
        qp, 16LL * N_ * M_, (long long)N_ * M_, M_,
        ctx, 16LL * M_ * DH_, (long long)M_ * DH_, DH_,
        at, (long long)N_ * D_, 64LL, D_,
        nullptr, dv, 16LL * N_, (long long)N_, M_);

    // final projection: attn [16384 x 1024] * Wo [1024 x 1024] + bo -> fp32 out
    gemm_bf16<false, false, true, false><<<gq, blk, 0, stream>>>(
        at, 0, 0, 1024, wob, 0, 0, 1024, d_out, 0, 0, 1024, bo, nullptr, 0, 0, 1024);
}
